// CoEncoderDynamicWeightedAvgPool1d_7275674600035
// MI455X (gfx1250) — matrix operand order repaired and checked by reading, not compiled, not run on hardware
//
#include <hip/hip_runtime.h>
#include <hip/hip_bf16.h>
#include <cstdint>

// ---------------- problem constants (match reference) ----------------
constexpr int Bc  = 4;      // batch
constexpr int Sc  = 2048;   // sequence
constexpr int HIDc = 1024;  // hidden
constexpr int Hc  = 16;     // query heads
constexpr int KVc = 4;      // kv heads
constexpr int Dc  = 64;     // head dim
constexpr int OUT_MINc = 32;
constexpr int OUT_MAXc = 8192;

// ---------------- vector types ----------------
typedef __attribute__((ext_vector_type(16))) __bf16 v16bf;
typedef __attribute__((ext_vector_type(8)))  __bf16 v8bf;
typedef __attribute__((ext_vector_type(8)))  float  v8f;

// ---------------- bf16 scalar helpers (RNE, used only in conversion pass) ----
__device__ __forceinline__ unsigned short f2bfbits(float f) {
    union { float f; unsigned u; } v; v.f = f;
    return (unsigned short)((v.u + 0x7FFFu + ((v.u >> 16) & 1u)) >> 16);
}
__device__ __forceinline__ float bfbits2f(unsigned short h) {
    union { unsigned u; float f; } v; v.u = ((unsigned)h) << 16; return v.f;
}

// ---------------- WMMA fragment loaders (contiguous 16B chunks per lane) ----
// A-matrix (16x32 bf16, MxK), ISA 7.12.2: lane(row)=lane&15, khalf=lane>>4:
//   elements 0..7  <- K = khalf*8 + [0,8)
//   elements 8..15 <- K = 16 + khalf*8 + [0,8)
__device__ __forceinline__ v16bf load_a_frag(const unsigned short* row_k0, int khalf) {
    const v8bf lo = *reinterpret_cast<const v8bf*>(row_k0 + khalf * 8);
    const v8bf hi = *reinterpret_cast<const v8bf*>(row_k0 + 16 + khalf * 8);
    return __builtin_shufflevector(lo, hi, 0, 1, 2, 3, 4, 5, 6, 7,
                                           8, 9, 10, 11, 12, 13, 14, 15);
}
// B-matrix (32x16 bf16, KxN): lane(col)=lane&15; elements 0..7 <- K = khalf*8 + e, 8..15 <- K = 16 + khalf*8 + (e-8).
// Source must be laid out K-contiguous for the lane's column (i.e. N x K).
__device__ __forceinline__ v16bf load_b_frag(const unsigned short* col_k0, int khalf) {
    const v8bf lo = *reinterpret_cast<const v8bf*>(col_k0 + khalf * 8);
    const v8bf hi = *reinterpret_cast<const v8bf*>(col_k0 + 16 + khalf * 8);
    return __builtin_shufflevector(lo, hi, 0, 1, 2, 3, 4, 5, 6, 7,
                                           8, 9, 10, 11, 12, 13, 14, 15);
}
__device__ __forceinline__ v8f wmma_bf16(v16bf a, v16bf b, v8f c) {
    return __builtin_amdgcn_wmma_f32_16x16x32_bf16(false, a, false, b,
                                                   (short)0, c, false, false);
}

// =====================================================================
// 0a) f32 -> bf16 bulk conversion (activations)
// =====================================================================
__global__ __launch_bounds__(256)
void conv_bf16(const float* __restrict__ src, unsigned short* __restrict__ dst, int n) {
    int i = blockIdx.x * 256 + threadIdx.x;
    if (i < n) dst[i] = f2bfbits(src[i]);
}

// =====================================================================
// 0b) W (KxN f32) -> Wt (NxK bf16): makes B-fragments K-contiguous per lane
// =====================================================================
__global__ __launch_bounds__(256)
void transpose_w_bf16(const float* __restrict__ W, unsigned short* __restrict__ Wt,
                      int K, int N) {
    int idx = blockIdx.x * 256 + threadIdx.x;   // over N*K, k fastest
    if (idx >= N * K) return;
    int k = idx % K, n = idx / K;
    Wt[idx] = f2bfbits(W[(size_t)k * N + n]);
}

// =====================================================================
// 1) GEMM: C(bf16, MxN) = A(bf16, MxK) @ Wt(bf16, NxK)^T
//    One 32x32 tile per wave (4 accumulators, 4 WMMAs / K-step),
//    4 waves per 128-thread block -> 64x64 block tile.
// =====================================================================
__global__ __launch_bounds__(128)
void gemm_bf16_wmma(const unsigned short* __restrict__ Abf,
                    const unsigned short* __restrict__ Wt,
                    unsigned short* __restrict__ C, int M, int N, int K) {
    const int lane  = threadIdx.x & 31;
    const int wave  = threadIdx.x >> 5;
    const int khalf = lane >> 4;
    const int nl    = lane & 15;
    const int tileM = blockIdx.y * 64 + (wave >> 1) * 32;
    const int tileN = blockIdx.x * 64 + (wave & 1) * 32;

    const unsigned short* a0 = Abf + (size_t)(tileM + nl) * K;
    const unsigned short* a1 = Abf + (size_t)(tileM + 16 + nl) * K;
    const unsigned short* b0 = Wt + (size_t)(tileN + nl) * K;
    const unsigned short* b1 = Wt + (size_t)(tileN + 16 + nl) * K;

    v8f c00 = {}, c01 = {}, c10 = {}, c11 = {};
    for (int k0 = 0; k0 < K; k0 += 32) {
        __builtin_prefetch(a0 + k0 + 128, 0, 3);
        __builtin_prefetch(b0 + k0 + 128, 0, 3);
        v16bf A0 = load_a_frag(a0 + k0, khalf);
        v16bf A1 = load_a_frag(a1 + k0, khalf);
        v16bf B0 = load_b_frag(b0 + k0, khalf);
        v16bf B1 = load_b_frag(b1 + k0, khalf);
        c00 = wmma_bf16(A0, B0, c00);
        c01 = wmma_bf16(A0, B1, c01);
        c10 = wmma_bf16(A1, B0, c10);
        c11 = wmma_bf16(A1, B1, c11);
    }
#pragma unroll
    for (int r = 0; r < 8; ++r) {
        int row0 = tileM + r + 8 * khalf;
        int row1 = row0 + 16;
        C[(size_t)row0 * N + tileN + nl]      = f2bfbits(c00[r]);
        C[(size_t)row0 * N + tileN + 16 + nl] = f2bfbits(c01[r]);
        C[(size_t)row1 * N + tileN + nl]      = f2bfbits(c10[r]);
        C[(size_t)row1 * N + tileN + 16 + nl] = f2bfbits(c11[r]);
    }
}

// =====================================================================
// 2) vw[b,h,t] = V[b,t,h>>2,:] . Wo[h*D:(h+1)*D]   (rank-1 o_proj collapse)
// =====================================================================
__global__ __launch_bounds__(256)
void vw_dot(const unsigned short* __restrict__ vbuf, const float* __restrict__ Wo,
            float* __restrict__ vwbuf) {
    int idx = blockIdx.x * 256 + threadIdx.x;          // over B*H*S
    int t = idx % Sc; int bh = idx / Sc;
    int h = bh % Hc;  int b  = bh / Hc;
    const unsigned short* vp = vbuf + ((size_t)(b * Sc + t) * KVc + (h >> 2)) * Dc;
    const float* wp = Wo + h * Dc;
    float acc = 0.f;
#pragma unroll 8
    for (int d = 0; d < Dc; ++d) acc += bfbits2f(vp[d]) * wp[d];
    vwbuf[idx] = acc;
}

// =====================================================================
// 3) Flash-style scalar attention score:
//    s[b,q] = sum_h softmax_k(QK^T/8) . vw[b,h,:]
//    Block = (b, qtile); 4 waves, wave w handles heads {w, w+4, w+8, w+12}.
//    Q.K^T per 16x16 tile = 2 x v_wmma_f32_16x16x32_bf16 (D=64); fragments
//    are contiguous b128 loads (K projection is d-major in memory).
// =====================================================================
__global__ __launch_bounds__(128)
void attn_score(const unsigned short* __restrict__ qbuf,
                const unsigned short* __restrict__ kbuf,
                const float* __restrict__ vwbuf,
                float* __restrict__ sbuf) {
    const int b     = blockIdx.y;
    const int qtile = blockIdx.x;
    const int lane  = threadIdx.x & 31;
    const int wave  = threadIdx.x >> 5;
    const int khalf = lane >> 4;
    const int nl    = lane & 15;

    __shared__ float lds_acc[4][16];

    float acc[8];
#pragma unroll
    for (int r = 0; r < 8; ++r) acc[r] = 0.f;

    for (int hh = 0; hh < 4; ++hh) {
        const int h   = wave + 4 * hh;
        const int kvh = h >> 2;

        const unsigned short* qp =
            qbuf + (size_t)(b * Sc + qtile * 16 + nl) * (Hc * Dc) + h * Dc;
        v16bf qa0 = load_a_frag(qp, khalf);        // d in [0,32)
        v16bf qa1 = load_a_frag(qp + 32, khalf);   // d in [32,64)

        float rm[8], rd[8], rn[8];
#pragma unroll
        for (int r = 0; r < 8; ++r) { rm[r] = -1e30f; rd[r] = 0.f; rn[r] = 0.f; }

        for (int kt = 0; kt < Sc / 16; ++kt) {
            const int kvtok = kt * 16 + nl;        // this lane's column token (N)
            const unsigned short* kp =
                kbuf + ((size_t)(b * Sc + kvtok) * KVc + kvh) * Dc;
            __builtin_prefetch(kp + 16 * KVc * Dc, 0, 3);
            v16bf kb0 = load_b_frag(kp, khalf);        // d in [0,32)
            v16bf kb1 = load_b_frag(kp + 32, khalf);   // d in [32,64)

            v8f c = {};
            c = wmma_bf16(qa0, kb0, c);
            c = wmma_bf16(qa1, kb1, c);
            const float vwv = vwbuf[((size_t)(b * Hc + h)) * Sc + kvtok];

            // online softmax; row m = r + 8*khalf spans the 16 lanes of a half
#pragma unroll
            for (int r = 0; r < 8; ++r) {
                float sc = c[r] * 0.125f;              // 1/sqrt(64)
                float tm = sc;
                for (int m = 1; m <= 8; m <<= 1)
                    tm = fmaxf(tm, __shfl_xor(tm, m, 32));
                float nm = fmaxf(rm[r], tm);
                float e0 = __expf(sc - nm);
                float es = e0, ev = e0 * vwv;
                for (int m = 1; m <= 8; m <<= 1) {
                    es += __shfl_xor(es, m, 32);
                    ev += __shfl_xor(ev, m, 32);
                }
                float f = __expf(rm[r] - nm);
                rd[r] = rd[r] * f + es;
                rn[r] = rn[r] * f + ev;
                rm[r] = nm;
            }
        }
#pragma unroll
        for (int r = 0; r < 8; ++r) acc[r] += rn[r] / rd[r];
    }

    if (nl == 0) {
#pragma unroll
        for (int r = 0; r < 8; ++r) lds_acc[wave][8 * khalf + r] = acc[r];
    }
    __syncthreads();
    if (threadIdx.x < 16) {
        float t = lds_acc[0][threadIdx.x] + lds_acc[1][threadIdx.x] +
                  lds_acc[2][threadIdx.x] + lds_acc[3][threadIdx.x];
        sbuf[(size_t)b * Sc + qtile * 16 + threadIdx.x] = t;
    }
}

// =====================================================================
// 4) sizes[b] = int(mean_q sigmoid(s1) * scale * (OUT_MAX-OUT_MIN) + OUT_MIN)
// =====================================================================
__global__ __launch_bounds__(256)
void sizes_kernel(const float* __restrict__ sbuf, const float* __restrict__ scale,
                  int* __restrict__ sizes_ws, int* __restrict__ out_sizes) {
    const int b = blockIdx.x;
    __shared__ float red[256];
    float acc = 0.f;
    for (int t = threadIdx.x; t < Sc; t += 256) {
        float s = sbuf[(size_t)b * Sc + t];
        acc += 1.f / (1.f + __expf(-s));
    }
    red[threadIdx.x] = acc;
    __syncthreads();
    for (int st = 128; st > 0; st >>= 1) {
        if (threadIdx.x < st) red[threadIdx.x] += red[threadIdx.x + st];
        __syncthreads();
    }
    if (threadIdx.x == 0) {
        float mean = red[0] / (float)Sc;
        int sz = (int)(mean * (*scale) * (float)(OUT_MAXc - OUT_MINc) + (float)OUT_MINc);
        if (sz < 1) sz = 1;
        sizes_ws[b]  = sz;
        out_sizes[b] = sz;
    }
}

// =====================================================================
// 5) w = sigmoid(s2)
// =====================================================================
__global__ __launch_bounds__(256)
void sigmoid_kernel(const float* __restrict__ s, float* __restrict__ w, int n) {
    int i = blockIdx.x * 256 + threadIdx.x;
    if (i < n) w[i] = 1.f / (1.f + __expf(-s[i]));
}

// =====================================================================
// 6) Gather-based weighted segment pooling (right-aligned, no atomics).
// =====================================================================
__global__ __launch_bounds__(256)
void pool_kernel(const float* __restrict__ x, const float* __restrict__ wbuf,
                 const int* __restrict__ sizes_ws, float* __restrict__ pooled,
                 float* __restrict__ maskf, int max_len) {
    const int b = blockIdx.y, j = blockIdx.x;
    int sz = sizes_ws[b];
    if (sz > max_len) sz = max_len;
    const int pad = max_len - sz;
    float* prow = pooled + ((size_t)b * max_len + j) * HIDc;
    if (j < pad) {
        for (int d = threadIdx.x; d < HIDc; d += 256) prow[d] = 0.f;
        if (threadIdx.x == 0) maskf[(size_t)b * max_len + j] = 0.f;
        return;
    }
    const int s  = j - pad;
    const int t0 = (int)(((long long)s * Sc) / sz);
    const int t1 = (int)(((long long)(s + 1) * Sc) / sz);
    float den = 0.f;
    for (int t = t0; t < t1; ++t) den += wbuf[(size_t)b * Sc + t];
    const float inv = 1.f / (den + 1e-8f);
    for (int d = threadIdx.x; d < HIDc; d += 256) {
        float num = 0.f;
        for (int t = t0; t < t1; ++t)
            num += wbuf[(size_t)b * Sc + t] * x[((size_t)b * Sc + t) * HIDc + d];
        prow[d] = num * inv;
    }
    if (threadIdx.x == 0) maskf[(size_t)b * max_len + j] = 1.f;
}

// =====================================================================
extern "C" void kernel_launch(void* const* d_in, const int* in_sizes, int n_in,
                              void* d_out, int out_size, void* d_ws, size_t ws_size,
                              hipStream_t stream) {
    (void)in_sizes; (void)n_in; (void)ws_size;
    const float* x     = (const float*)d_in[0];
    const float* Wq1   = (const float*)d_in[1];
    const float* Wk1   = (const float*)d_in[2];
    const float* Wv1   = (const float*)d_in[3];
    const float* Wo1   = (const float*)d_in[4];
    const float* Wq2   = (const float*)d_in[5];
    const float* Wk2   = (const float*)d_in[6];
    const float* Wv2   = (const float*)d_in[7];
    const float* Wo2   = (const float*)d_in[8];
    const float* scale = (const float*)d_in[9];

    // out_size = B*(max_len*(HID+1) + 1)  ->  recover max_len on host
    int max_len = (out_size / Bc - 1) / (HIDc + 1);
    if (max_len < 1) max_len = 1;

    // ---- workspace carve (bf16 activations/weights, f32 scalars) ----
    const size_t nX  = (size_t)Bc * Sc * HIDc;       // 8.4M
    const size_t nQ  = (size_t)Bc * Sc * Hc * Dc;    // 8.4M
    const size_t nKV = (size_t)Bc * Sc * KVc * Dc;   // 2.1M
    unsigned short* xbf  = (unsigned short*)d_ws;
    unsigned short* qbuf = xbf  + nX;
    unsigned short* kbuf = qbuf + nQ;
    unsigned short* vbuf = kbuf + nKV;
    unsigned short* wqt  = vbuf + nKV;               // 1024*1024
    unsigned short* wkt  = wqt + (size_t)HIDc * (Hc * Dc) / 1;  // actually N*K
    unsigned short* wvt  = wkt + (size_t)(KVc * Dc) * HIDc;
    float* vwbuf = (float*)(wvt + (size_t)(KVc * Dc) * HIDc);
    float* sbuf  = vwbuf + (size_t)Bc * Hc * Sc;
    float* wbuf  = sbuf  + (size_t)Bc * Sc;
    int*   sizes_ws = (int*)(wbuf + (size_t)Bc * Sc);

    float* pooled = (float*)d_out;
    float* maskf  = pooled + (size_t)Bc * max_len * HIDc;
    int*   out_sz = (int*)(maskf + (size_t)Bc * max_len);

    const int M = Bc * Sc;
    const dim3 gQ(Hc * Dc / 64, M / 64), gKV(KVc * Dc / 64, M / 64);

    // ---- one-time activation conversion ----
    conv_bf16<<<(int)(nX / 256), 256, 0, stream>>>(x, xbf, (int)nX);

    // -------- pass 1: size-estimation attention --------
    transpose_w_bf16<<<(HIDc * Hc * Dc) / 256, 256, 0, stream>>>(Wq1, wqt, HIDc, Hc * Dc);
    transpose_w_bf16<<<(HIDc * KVc * Dc) / 256, 256, 0, stream>>>(Wk1, wkt, HIDc, KVc * Dc);
    transpose_w_bf16<<<(HIDc * KVc * Dc) / 256, 256, 0, stream>>>(Wv1, wvt, HIDc, KVc * Dc);
    gemm_bf16_wmma<<<gQ, 128, 0, stream>>>(xbf, wqt, qbuf, M, Hc * Dc, HIDc);
    gemm_bf16_wmma<<<gKV, 128, 0, stream>>>(xbf, wkt, kbuf, M, KVc * Dc, HIDc);
    gemm_bf16_wmma<<<gKV, 128, 0, stream>>>(xbf, wvt, vbuf, M, KVc * Dc, HIDc);
    vw_dot<<<(Bc * Hc * Sc) / 256, 256, 0, stream>>>(vbuf, Wo1, vwbuf);
    attn_score<<<dim3(Sc / 16, Bc), 128, 0, stream>>>(qbuf, kbuf, vwbuf, sbuf);
    sizes_kernel<<<Bc, 256, 0, stream>>>(sbuf, scale, sizes_ws, out_sz);

    // -------- pass 2: weighted-pooling attention (reuse buffers) --------
    transpose_w_bf16<<<(HIDc * Hc * Dc) / 256, 256, 0, stream>>>(Wq2, wqt, HIDc, Hc * Dc);
    transpose_w_bf16<<<(HIDc * KVc * Dc) / 256, 256, 0, stream>>>(Wk2, wkt, HIDc, KVc * Dc);
    transpose_w_bf16<<<(HIDc * KVc * Dc) / 256, 256, 0, stream>>>(Wv2, wvt, HIDc, KVc * Dc);
    gemm_bf16_wmma<<<gQ, 128, 0, stream>>>(xbf, wqt, qbuf, M, Hc * Dc, HIDc);
    gemm_bf16_wmma<<<gKV, 128, 0, stream>>>(xbf, wkt, kbuf, M, KVc * Dc, HIDc);
    gemm_bf16_wmma<<<gKV, 128, 0, stream>>>(xbf, wvt, vbuf, M, KVc * Dc, HIDc);
    vw_dot<<<(Bc * Hc * Sc) / 256, 256, 0, stream>>>(vbuf, Wo2, vwbuf);
    attn_score<<<dim3(Sc / 16, Bc), 128, 0, stream>>>(qbuf, kbuf, vwbuf, sbuf);
    sigmoid_kernel<<<(Bc * Sc) / 256, 256, 0, stream>>>(sbuf, wbuf, Bc * Sc);

    // -------- dynamic-size segment pooling --------
    pool_kernel<<<dim3(max_len, Bc), 256, 0, stream>>>(x, wbuf, sizes_ws,
                                                       pooled, maskf, max_len);
}
